// Encoding_7954279432349
// MI455X (gfx1250) — matrix operand order repaired and checked by reading, not compiled, not run on hardware
//
#include <hip/hip_runtime.h>

// ---------------- problem constants ----------------
#define B_  32
#define C_  512
#define N_  4096        // H*W
#define K_  32
#define CH_ 8           // column chunks per batch -> grid = B_*CH_ = 256 WGs
#define NCHUNK (N_/CH_) // 512 columns per WG
#define TN  64          // columns per LDS tile
#define NTILES (NCHUNK/TN) // 8

typedef _Float16 h16;
typedef __attribute__((ext_vector_type(8)))  _Float16 v8h;
typedef __attribute__((ext_vector_type(16))) _Float16 v16h;
typedef __attribute__((ext_vector_type(8)))  float    v8f;
typedef __attribute__((ext_vector_type(4)))  float    v4f;
typedef __attribute__((ext_vector_type(4)))  int      v4i;

#define AS1 __attribute__((address_space(1)))
#define AS3 __attribute__((address_space(3)))

// ---- async global->LDS copy (16B per lane), ASYNCcnt-tracked ----
static __device__ __forceinline__ void async_ld16B(const void* g, void* l) {
#if __has_builtin(__builtin_amdgcn_global_load_async_to_lds_b128)
  __builtin_amdgcn_global_load_async_to_lds_b128((AS1 v4i*)g, (AS3 v4i*)l, 0, 0);
#else
  asm volatile("global_load_async_to_lds_b128 %0, %1, off"
               :: "v"((unsigned)(unsigned long long)l),
                  "v"((unsigned long long)g)
               : "memory");
#endif
}

static __device__ __forceinline__ void wait_async0() {
#if __has_builtin(__builtin_amdgcn_s_wait_asynccnt)
  __builtin_amdgcn_s_wait_asynccnt(0);
#else
  asm volatile("s_wait_asynccnt 0x0" ::: "memory");
#endif
}

// two contiguous 16B LDS loads -> one v16h WMMA operand (8 VGPRs)
static __device__ __forceinline__ v16h ld16(const h16* p0, const h16* p1) {
  v8h a = *(const v8h*)p0;
  v8h b = *(const v8h*)p1;
  v16h r;
#pragma unroll
  for (int i = 0; i < 8; ++i) { r[i] = a[i]; r[8 + i] = b[i]; }
  return r;
}

static __device__ __forceinline__ v8f wmma16(v16h a, v16h b, v8f c) {
  // D = A(16x32 f16) * B(32x16 f16) + C(16x16 f32)
  return __builtin_amdgcn_wmma_f32_16x16x32_f16(false, a, false, b, (short)0, c,
                                                false, false);
}

// ---- LDS layout (dynamic shared, 307840 B < 320 KB/WGP) ----
#define OFF_STG  0                 // f32 [C_][TN]     131072 B (async staging)
#define OFF_XL   131072            // h16 [C_][TN]      65536 B
#define OFF_XT   196608            // h16 [TN][C_]      65536 B
#define OFF_CW   262144            // h16 [K_][C_]      32768 B
#define OFF_XC   294912            // f32 [K_][TN]       8192 B (also x2 scratch)
#define OFF_W    303104            // h16 [K_][TN]       4096 B
#define OFF_X2   307200            // f32 [TN]            256 B
#define OFF_C2   307456            // f32 [K_]            128 B
#define OFF_SC   307584            // f32 [K_]            128 B
#define OFF_WS   307712            // f32 [K_]            128 B
#define SMEM_BYTES 307840

// issue async loads for one 64-column tile: each thread owns the same
// stage slots that it later converts (per-wave ASYNCcnt wait is sufficient)
static __device__ __forceinline__ void issue_tile_async(
    const float* __restrict__ x, long long xbase, int n0, int tid, float* stage) {
  const int j4 = (tid & 15) * 4;                 // 4 floats = 16 B
  const float* src = x + xbase + n0 + j4;
  float* dst = stage + j4;
#pragma unroll 4
  for (int i = 0; i < 32; ++i) {
    const int c = (tid >> 4) + (i << 4);         // row 0..511
    async_ld16B(src + (long long)c * N_, dst + c * TN);
  }
}

__global__ __launch_bounds__(256) void enc_stage1(
    const float* __restrict__ x,     // [B_, C_, N_]
    const float* __restrict__ cw,    // [K_, C_]
    const float* __restrict__ scale, // [K_]
    float* __restrict__ part,        // [B_*CH_, K_, C_]
    float* __restrict__ wsp)         // [B_*CH_, K_]
{
  extern __shared__ char smem[];
  float* stg = (float*)(smem + OFF_STG);
  h16*   xl  = (h16*)  (smem + OFF_XL);
  h16*   xt  = (h16*)  (smem + OFF_XT);
  h16*   cwS = (h16*)  (smem + OFF_CW);
  float* xcS = (float*)(smem + OFF_XC);
  h16*   wS  = (h16*)  (smem + OFF_W);
  float* x2S = (float*)(smem + OFF_X2);
  float* c2S = (float*)(smem + OFF_C2);
  float* scS = (float*)(smem + OFF_SC);
  float* wsA = (float*)(smem + OFF_WS);

  const int tid  = threadIdx.x;
  const int lane = tid & 31;
  const int wv   = tid >> 5;          // wave 0..7 (wave32)
  const int hi   = lane >> 4;         // lane half
  const int l15  = lane & 15;

  const int wg    = blockIdx.x;       // 0..255
  const int b     = wg / CH_;
  const int chunk = wg % CH_;
  const long long xbase = (long long)b * C_ * N_ + (long long)chunk * NCHUNK;

  // kick off tile 0 fetch immediately (overlaps codeword setup below)
  issue_tile_async(x, xbase, 0, tid, stg);

  // ---------------- once per WG: codewords -> LDS f16, c2, scale ----------------
  for (int i = tid; i < K_ * C_ / 4; i += 256) {
    v4f g = ((const v4f*)cw)[i];
    h16* d = cwS + i * 4;
    d[0] = (h16)g.x; d[1] = (h16)g.y; d[2] = (h16)g.z; d[3] = (h16)g.w;
  }
  if (tid < K_) {
    float s = 0.f;
    const float* r = cw + tid * C_;
#pragma unroll 8
    for (int c = 0; c < C_; ++c) s += r[c] * r[c];
    c2S[tid] = s;
    scS[tid] = scale[tid];
    wsA[tid] = 0.f;
  }
  __syncthreads();

  // wx accumulators: this wave owns rows [kt*16,+16), cols [cg*128,+128)
  const int kt = wv & 1;
  const int cg = wv >> 1;             // 0..3
  v8f acc[8];
#pragma unroll
  for (int i = 0; i < 8; ++i) acc[i] = (v8f){0.f,0.f,0.f,0.f,0.f,0.f,0.f,0.f};

  // ---------------- main loop over 64-column tiles ----------------
  for (int t = 0; t < NTILES; ++t) {
    // ---- tile t data has landed in stage once our ASYNCcnt drains ----
    wait_async0();

    // ---- convert stage f32 -> LDS f16 (both layouts) + f32 x2 partials ----
    float p0 = 0.f, p1 = 0.f, p2 = 0.f, p3 = 0.f;
    const int j4 = (tid & 15) * 4;    // column group (fixed per thread)
#pragma unroll 4
    for (int i = 0; i < 32; ++i) {
      const int c = (tid >> 4) + (i << 4);              // row 0..511
      v4f g = *(const v4f*)(stg + c * TN + j4);
      p0 += g.x * g.x; p1 += g.y * g.y; p2 += g.z * g.z; p3 += g.w * g.w;
      h16* dl = xl + c * TN + j4;                       // [C][TN]
      dl[0] = (h16)g.x; dl[1] = (h16)g.y; dl[2] = (h16)g.z; dl[3] = (h16)g.w;
      xt[(j4 + 0) * C_ + c] = (h16)g.x;                 // [TN][C]
      xt[(j4 + 1) * C_ + c] = (h16)g.y;
      xt[(j4 + 2) * C_ + c] = (h16)g.z;
      xt[(j4 + 3) * C_ + c] = (h16)g.w;
    }
    // stage slots of this thread are consumed -> prefetch next tile into them
    if (t + 1 < NTILES)
      issue_tile_async(x, xbase, (t + 1) * TN, tid, stg);

    // deterministic x2 reduction via scratch (xcS region is free here)
    float* scr = xcS;
    scr[tid * 4 + 0] = p0; scr[tid * 4 + 1] = p1;
    scr[tid * 4 + 2] = p2; scr[tid * 4 + 3] = p3;
    __syncthreads();
    if (tid < TN) {
      const int g = tid >> 2, m = tid & 3;
      float s = 0.f;
#pragma unroll
      for (int q = 0; q < 16; ++q) s += scr[(g + (q << 4)) * 4 + m];
      x2S[tid] = s;
    }
    __syncthreads();

    // ---- GEMM1: xc[16k x 16n] per wave; contraction over C in steps of 32 ----
    {
      const int nt = wv >> 1;                           // 0..3 (4 x 16 = 64 cols)
      v8f c = (v8f){0.f,0.f,0.f,0.f,0.f,0.f,0.f,0.f};
      const h16* ap = cwS + (kt * 16 + l15) * C_ + hi * 8;   // A: cw row k
      const h16* bp = xt  + (nt * 16 + l15) * C_ + hi * 8;   // B: x^T col n
#pragma unroll 4
      for (int cb = 0; cb < C_; cb += 32) {
        v16h a  = ld16(ap + cb, ap + cb + 16);
        v16h bm = ld16(bp + cb, bp + cb + 16);
        c = wmma16(a, bm, c);
      }
#pragma unroll
      for (int r = 0; r < 8; ++r)
        xcS[(kt * 16 + r + hi * 8) * TN + nt * 16 + l15] = c[r];
    }
    __syncthreads();

    // ---- softmax over K per column (f32), store w as f16 ----
    if (tid < TN) {
      const int j = tid;
      const float x2v = x2S[j];
      float m = -3.4e38f;
#pragma unroll
      for (int k = 0; k < K_; ++k) {
        float dv = scS[k] * (x2v - 2.f * xcS[k * TN + j] + c2S[k]);
        m = fmaxf(m, dv);
      }
      float s = 0.f;
#pragma unroll
      for (int k = 0; k < K_; ++k) {
        float dv = scS[k] * (x2v - 2.f * xcS[k * TN + j] + c2S[k]);
        float e = __expf(dv - m);
        xcS[k * TN + j] = e;                 // overwrite in place
        s += e;
      }
      const float inv = 1.f / s;
#pragma unroll
      for (int k = 0; k < K_; ++k)
        wS[k * TN + j] = (h16)(xcS[k * TN + j] * inv);
    }
    __syncthreads();

    // ---- wsum (single writer per k -> deterministic) ----
    if (tid < K_) {
      float s = 0.f;
#pragma unroll 8
      for (int j = 0; j < TN; ++j) s += (float)wS[tid * TN + j];
      wsA[tid] += s;
    }

    // ---- GEMM2: wx += w[16k x 64n] @ x[64n x 128c]; contraction over n ----
    {
      const h16* ap = wS + (kt * 16 + l15) * TN + hi * 8;   // A: w row k
#pragma unroll
      for (int cc = 0; cc < 8; ++cc) {
        const int ct = cg * 8 + cc;
        const h16* bp = xl + (ct * 16 + l15) * TN + hi * 8;  // B: x row c
        v16h a0 = ld16(ap + 0,  ap + 16);   // n = 0..31
        v16h b0 = ld16(bp + 0,  bp + 16);
        acc[cc] = wmma16(a0, b0, acc[cc]);
        v16h a1 = ld16(ap + 32, ap + 48);   // n = 32..63
        v16h b1 = ld16(bp + 32, bp + 48);
        acc[cc] = wmma16(a1, b1, acc[cc]);
      }
    }
    __syncthreads();  // protect xl/xt/wS before next tile's conversion
  }

  // ---------------- write per-WG partials (fully written, no zero-init needed) ----
  {
    float* pb = part + (long long)wg * (K_ * C_);
#pragma unroll
    for (int cc = 0; cc < 8; ++cc) {
      const int ct = cg * 8 + cc;
#pragma unroll
      for (int r = 0; r < 8; ++r)
        pb[(kt * 16 + r + hi * 8) * C_ + ct * 16 + l15] = acc[cc][r];
    }
  }
  if (tid < K_) wsp[wg * K_ + tid] = wsA[tid];
}

// ---------------- stage 2: reduce partials, apply -wsum*cw ----------------
__global__ __launch_bounds__(256) void enc_stage2(
    const float* __restrict__ part, const float* __restrict__ wsp,
    const float* __restrict__ cw, float* __restrict__ out)
{
  const int idx = blockIdx.x * 256 + threadIdx.x;   // over B*K*C
  if (idx >= B_ * K_ * C_) return;
  const int c = idx % C_;
  const int k = (idx / C_) % K_;
  const int b = idx / (K_ * C_);
  float s = 0.f, ws = 0.f;
#pragma unroll
  for (int ch = 0; ch < CH_; ++ch) {
    const int g = b * CH_ + ch;
    s  += part[(long long)g * (K_ * C_) + k * C_ + c];
    ws += wsp[g * K_ + k];
  }
  out[idx] = s - ws * cw[k * C_ + c];
}

extern "C" void kernel_launch(void* const* d_in, const int* in_sizes, int n_in,
                              void* d_out, int out_size, void* d_ws, size_t ws_size,
                              hipStream_t stream) {
  const float* x  = (const float*)d_in[0];   // [B, C, H, W] f32
  const float* cw = (const float*)d_in[1];   // [K, C] f32
  const float* sc = (const float*)d_in[2];   // [K] f32

  float* part = (float*)d_ws;                        // 256 * K * C f32 = 16 MB
  float* wsp  = part + (size_t)B_ * CH_ * K_ * C_;   // 256 * K f32

  enc_stage1<<<B_ * CH_, 256, SMEM_BYTES, stream>>>(x, cw, sc, part, wsp);
  enc_stage2<<<(B_ * K_ * C_ + 255) / 256, 256, 0, stream>>>(part, wsp, cw,
                                                             (float*)d_out);
}
